// GraphSAGEModel_66408784331010
// MI455X (gfx1250) — matrix operand order repaired and checked by reading, not compiled, not run on hardware
//
#include <hip/hip_runtime.h>
#include <hip/hip_bf16.h>
#include <math.h>

// ---------------------------------------------------------------------------
// GraphSAGE inference for MI455X (gfx1250, wave32, WMMA 16x16x32 f16).
// h (25.6MB) + edge_index (12.8MB) are L2-resident; dense GEMMs use
// v_wmma_f32_16x16x32_f16 with b128-vectorized fragment loads; weights are
// pre-transposed to f16 [N][K] so B fragments are 2x global_load_b128.
// ---------------------------------------------------------------------------

typedef __attribute__((ext_vector_type(16))) _Float16 v16h;
typedef __attribute__((ext_vector_type(8)))  _Float16 v8h;
typedef __attribute__((ext_vector_type(8)))  float    v8f;

#define GN_NODES 100000
#define GN_EDGES 1600000
#define GHID 64
#define GNODE_DIM 128
#define WAVES_PER_BLOCK 8

static __device__ __forceinline__ v8f wmma16(v16h a, v16h b, v8f c) {
  // (neg_a, A, neg_b, B, c_mod, C, reuse_a, reuse_b)
  return __builtin_amdgcn_wmma_f32_16x16x32_f16(false, a, false, b, (short)0, c,
                                                false, false);
}

// A fragment (16x32, row-major f32 source -> f16), vectorized: 4x b128 loads.
// Per ISA 7.12.2: lane = M; VGPR0-3 hold K = kBase + 8*half + 0..7,
// VGPR4-7 hold K = kBase + 16 + 8*half + 0..7.
static __device__ __forceinline__ v16h load_a_frag(const float* __restrict__ rowPtr,
                                                   int kBase, int half, float scale) {
  const float* p = rowPtr + kBase + 8 * half;
  float4 f0 = *(const float4*)(p);
  float4 f1 = *(const float4*)(p + 4);
  float4 f2 = *(const float4*)(p + 16);
  float4 f3 = *(const float4*)(p + 20);
  v16h a;
  a[0]  = (_Float16)(f0.x * scale); a[1]  = (_Float16)(f0.y * scale);
  a[2]  = (_Float16)(f0.z * scale); a[3]  = (_Float16)(f0.w * scale);
  a[4]  = (_Float16)(f1.x * scale); a[5]  = (_Float16)(f1.y * scale);
  a[6]  = (_Float16)(f1.z * scale); a[7]  = (_Float16)(f1.w * scale);
  a[8]  = (_Float16)(f2.x * scale); a[9]  = (_Float16)(f2.y * scale);
  a[10] = (_Float16)(f2.z * scale); a[11] = (_Float16)(f2.w * scale);
  a[12] = (_Float16)(f3.x * scale); a[13] = (_Float16)(f3.y * scale);
  a[14] = (_Float16)(f3.z * scale); a[15] = (_Float16)(f3.w * scale);
  return a;
}

// B fragment (32x16) from pre-transposed f16 weights wT[N][K]: lane = N;
// elements 0-7 hold K=kBase+8*half+0..7, elements 8-15 hold K=kBase+16+8*half+0..7.
// 2x8 contiguous f16 per lane -> 2x b128 loads.
static __device__ __forceinline__ v16h load_b_frag(const _Float16* __restrict__ wT, int K,
                                                   int kBase, int nBase, int lane) {
  int n = lane & 15, half = lane >> 4;
  const _Float16* p = wT + (size_t)(nBase + n) * K + kBase + 8 * half;
  v8h lo = *(const v8h*)(p);
  v8h hi = *(const v8h*)(p + 16);
  v16h b;
#pragma unroll
  for (int e = 0; e < 8; ++e) { b[e] = lo[e]; b[e + 8] = hi[e]; }
  return b;
}

// C/D fragment store: VGPR r holds row (r + 8*half), column = lane&15.
static __device__ __forceinline__ void store_c_bias_relu(float* __restrict__ out, int ldN,
                                                         int rowBase, int nBase,
                                                         const float* __restrict__ bias,
                                                         v8f c, int lane) {
  int n = lane & 15, half = lane >> 4;
  float bv = bias[nBase + n];
#pragma unroll
  for (int r = 0; r < 8; ++r) {
    float v = c[r] + bv;
    out[(size_t)(rowBase + r + 8 * half) * ldN + nBase + n] = v > 0.f ? v : 0.f;
  }
}

// ---------------- small utility kernels ----------------

__global__ void zero_kernel(float* __restrict__ p, long n) {
  long i = (long)blockIdx.x * blockDim.x + threadIdx.x;
  if (i < n) p[i] = 0.f;
}

__global__ void zero4_kernel(float4* __restrict__ p, long n4) {
  long i = (long)blockIdx.x * blockDim.x + threadIdx.x;
  if (i < n4) p[i] = make_float4(0.f, 0.f, 0.f, 0.f);
}

__global__ void copy_kernel(const float* __restrict__ in, float* __restrict__ out, long n) {
  long i = (long)blockIdx.x * blockDim.x + threadIdx.x;
  if (i < n) out[i] = in[i];
}

// transpose + convert weights: src f32 [K,N] -> dst f16 [N,K]
__global__ void wprep_kernel(const float* __restrict__ src, _Float16* __restrict__ dst,
                             int K, int N) {
  int i = blockIdx.x * blockDim.x + threadIdx.x;
  if (i < K * N) {
    int k = i / N, n = i - k * N;
    dst[(size_t)n * K + k] = (_Float16)src[i];
  }
}

__global__ void deg_kernel(const int* __restrict__ dst, float* __restrict__ deg, int E) {
  int e = blockIdx.x * blockDim.x + threadIdx.x;
  if (e < E) atomicAdd(&deg[dst[e]], 1.0f);
}

__global__ void inv_kernel(float* __restrict__ d, int n) {
  int i = blockIdx.x * blockDim.x + threadIdx.x;
  if (i < n) {
    float v = d[i];
    d[i] = (v > 0.f) ? (1.0f / v) : 0.0f;
  }
}

// scatter-add h[src] rows into agg[dst] rows (mean normalization applied later)
__global__ void scatter_kernel(const int* __restrict__ srcIdx, const int* __restrict__ dstIdx,
                               const float* __restrict__ hin, float* __restrict__ agg,
                               long total) {
  long t = (long)blockIdx.x * blockDim.x + threadIdx.x;
  if (t >= total) return;
  int e = (int)(t >> 6);     // edge
  int f = (int)(t & 63);     // feature
  int s = srcIdx[e];
  int d = dstIdx[e];
  atomicAdd(&agg[(size_t)d * GHID + f], hin[(size_t)s * GHID + f]);
}

// ---------------- WMMA GEMM kernels ----------------

// h = relu(x @ W + b), x:[N,128]. One 16-row tile per wave.
__global__ void encoder_kernel(const float* __restrict__ x, const _Float16* __restrict__ wT,
                               const float* __restrict__ bias, float* __restrict__ h,
                               int nTiles) {
  int wave = threadIdx.x >> 5, lane = threadIdx.x & 31;
  int tile = blockIdx.x * (blockDim.x >> 5) + wave;
  if (tile >= nTiles) return;               // wave-uniform: EXEC all-ones for WMMA
  int rowBase = tile * 16;
  int m = lane & 15, half = lane >> 4;
  const float* rowPtr = x + (size_t)(rowBase + m) * GNODE_DIM;
  v16h a[4];
#pragma unroll
  for (int kb = 0; kb < 4; ++kb) a[kb] = load_a_frag(rowPtr, kb * 32, half, 1.0f);
#pragma unroll
  for (int nt = 0; nt < 4; ++nt) {
    v8f acc = {};
#pragma unroll
    for (int kb = 0; kb < 4; ++kb)
      acc = wmma16(a[kb], load_b_frag(wT, GNODE_DIM, kb * 32, nt * 16, lane), acc);
    store_c_bias_relu(h, GHID, rowBase, nt * 16, bias, acc, lane);
  }
}

// h_new = relu((agg*inv_deg) @ lw + lb + h @ rw), computed in-place into agg.
// Safe: each wave reads its full 16-row A range before any store.
__global__ void layer_kernel(float* __restrict__ agg, const float* __restrict__ hin,
                             const _Float16* __restrict__ lwT, const float* __restrict__ lb,
                             const _Float16* __restrict__ rwT, const float* __restrict__ invdeg,
                             int nTiles) {
  int wave = threadIdx.x >> 5, lane = threadIdx.x & 31;
  int tile = blockIdx.x * (blockDim.x >> 5) + wave;
  if (tile >= nTiles) return;
  int rowBase = tile * 16;
  int m = lane & 15, half = lane >> 4;
  int row = rowBase + m;
  float sc = invdeg[row];                    // mean normalization fused into A load
  const float* aRow = agg + (size_t)row * GHID;
  const float* hRow = hin + (size_t)row * GHID;
  v16h aAgg[2], aH[2];
#pragma unroll
  for (int kb = 0; kb < 2; ++kb) {
    aAgg[kb] = load_a_frag(aRow, kb * 32, half, sc);
    aH[kb]   = load_a_frag(hRow, kb * 32, half, 1.0f);
  }
  v8f acc[4] = {};
#pragma unroll
  for (int nt = 0; nt < 4; ++nt) {
#pragma unroll
    for (int kb = 0; kb < 2; ++kb) {
      acc[nt] = wmma16(aAgg[kb], load_b_frag(lwT, GHID, kb * 32, nt * 16, lane), acc[nt]);
      acc[nt] = wmma16(aH[kb],   load_b_frag(rwT, GHID, kb * 32, nt * 16, lane), acc[nt]);
    }
  }
#pragma unroll
  for (int nt = 0; nt < 4; ++nt)
    store_c_bias_relu(agg, GHID, rowBase, nt * 16, lb, acc[nt], lane);
}

// Edge head: ef = [h[src], h[dst]] (128) -> relu(ef@W1+b1) (64) -> sigmoid(.@w2+b2)
// 16 edges per wave via WMMA; hid tile staged in LDS for the 64->1 dot.
__global__ void edge_head_kernel(const int* __restrict__ srcIdx, const int* __restrict__ dstIdx,
                                 const float* __restrict__ h,
                                 const _Float16* __restrict__ w1T, const float* __restrict__ b1,
                                 const float* __restrict__ w2, const float* __restrict__ b2,
                                 float* __restrict__ scores, float* __restrict__ sumOut,
                                 int eTiles) {
  __shared__ float hidLds[WAVES_PER_BLOCK][16][68];   // pad 64->68: conflict-free rows
  __shared__ float blockSum;
  if (threadIdx.x == 0) blockSum = 0.f;
  __syncthreads();

  int wave = threadIdx.x >> 5, lane = threadIdx.x & 31;
  int tile = blockIdx.x * WAVES_PER_BLOCK + wave;
  bool active = tile < eTiles;
  int tt = active ? tile : 0;                // redundant compute keeps barriers uniform
  int m = lane & 15, half = lane >> 4;
  int edge = tt * 16 + m;
  const float* sRow = h + (size_t)srcIdx[edge] * GHID;
  const float* dRow = h + (size_t)dstIdx[edge] * GHID;
  v16h a[4];
  a[0] = load_a_frag(sRow, 0,  half, 1.0f);  // K 0..63   = h[src]
  a[1] = load_a_frag(sRow, 32, half, 1.0f);
  a[2] = load_a_frag(dRow, 0,  half, 1.0f);  // K 64..127 = h[dst]
  a[3] = load_a_frag(dRow, 32, half, 1.0f);
  v8f acc[4] = {};
#pragma unroll
  for (int nt = 0; nt < 4; ++nt)
#pragma unroll
    for (int kb = 0; kb < 4; ++kb)
      acc[nt] = wmma16(a[kb], load_b_frag(w1T, GNODE_DIM, kb * 32, nt * 16, lane), acc[nt]);

#pragma unroll
  for (int nt = 0; nt < 4; ++nt) {
    float bv = b1[nt * 16 + m];
#pragma unroll
    for (int r = 0; r < 8; ++r) {
      float v = acc[nt][r] + bv;
      hidLds[wave][r + 8 * half][nt * 16 + m] = v > 0.f ? v : 0.f;
    }
  }
  __syncthreads();

  if (lane < 16) {
    float dot = 0.f;
#pragma unroll
    for (int k = 0; k < GHID; ++k) dot += hidLds[wave][lane][k] * w2[k];
    dot += b2[0];
    float s = 1.0f / (1.0f + expf(-dot));
    if (active) {
      scores[tt * 16 + lane] = s;
      atomicAdd(&blockSum, s);
    }
  }
  __syncthreads();
  if (threadIdx.x == 0)
    atomicAdd(sumOut, blockSum * (1.0f / (float)GN_EDGES));
}

// ---------------------------------------------------------------------------

extern "C" void kernel_launch(void* const* d_in, const int* in_sizes, int n_in,
                              void* d_out, int out_size, void* d_ws, size_t ws_size,
                              hipStream_t stream) {
  (void)in_sizes; (void)n_in; (void)out_size; (void)ws_size;

  const float* x          = (const float*)d_in[0];
  const int*   ei         = (const int*)d_in[1];
  const float* enc_node_w = (const float*)d_in[3];
  const float* enc_node_b = (const float*)d_in[4];
  const float* conv_lw    = (const float*)d_in[7];
  const float* conv_lb    = (const float*)d_in[8];
  const float* conv_rw    = (const float*)d_in[9];
  const float* eh_w1      = (const float*)d_in[10];
  const float* eh_b1      = (const float*)d_in[11];
  const float* eh_w2      = (const float*)d_in[12];
  const float* eh_b2      = (const float*)d_in[13];
  const int* src = ei;
  const int* dst = ei + GN_EDGES;

  // workspace: ping-pong node features (f32) + inv_deg + transposed f16 weights
  float* P      = (float*)d_ws;
  float* Q      = P + (size_t)GN_NODES * GHID;
  float* invdeg = Q + (size_t)GN_NODES * GHID;
  _Float16* wf16 = (_Float16*)(invdeg + GN_NODES);   // 16B-aligned (offset % 16 == 0)
  _Float16* wnT  = wf16;                              // [64][128]
  _Float16* lwT  = wnT + 64 * 128;                    // 2 x [64][64]
  _Float16* rwT  = lwT + 2 * 64 * 64;                 // 2 x [64][64]
  _Float16* w1T  = rwT + 2 * 64 * 64;                 // [64][128]

  // output layout: edge_scores[E], path_scores[1], h[N*64]
  float* scores  = (float*)d_out;
  float* pathSum = scores + GN_EDGES;
  float* hOut    = pathSum + 1;

  const int nodeTiles  = GN_NODES / 16;                                  // 6250
  const int nodeBlocks = (nodeTiles + WAVES_PER_BLOCK - 1) / WAVES_PER_BLOCK;
  const int eTiles     = GN_EDGES / 16;                                  // 100000
  const int eBlocks    = (eTiles + WAVES_PER_BLOCK - 1) / WAVES_PER_BLOCK;

  // weight prep: transpose+convert to f16 [N][K]
  wprep_kernel<<<(128 * 64 + 255) / 256, 256, 0, stream>>>(enc_node_w, wnT, 128, 64);
  wprep_kernel<<<(64 * 64 + 255) / 256, 256, 0, stream>>>(conv_lw, lwT, 64, 64);
  wprep_kernel<<<(64 * 64 + 255) / 256, 256, 0, stream>>>(conv_lw + 64 * 64, lwT + 64 * 64, 64, 64);
  wprep_kernel<<<(64 * 64 + 255) / 256, 256, 0, stream>>>(conv_rw, rwT, 64, 64);
  wprep_kernel<<<(64 * 64 + 255) / 256, 256, 0, stream>>>(conv_rw + 64 * 64, rwT + 64 * 64, 64, 64);
  wprep_kernel<<<(128 * 64 + 255) / 256, 256, 0, stream>>>(eh_w1, w1T, 128, 64);

  // degree -> inv_deg
  zero_kernel<<<(GN_NODES + 255) / 256, 256, 0, stream>>>(invdeg, (long)GN_NODES);
  deg_kernel<<<(GN_EDGES + 255) / 256, 256, 0, stream>>>(dst, invdeg, GN_EDGES);
  inv_kernel<<<(GN_NODES + 255) / 256, 256, 0, stream>>>(invdeg, GN_NODES);

  // node encoder
  encoder_kernel<<<nodeBlocks, 256, 0, stream>>>(x, wnT, enc_node_b, P, nodeTiles);

  // two SAGE layers, ping-pong P<->Q
  float* cur = P;
  float* oth = Q;
  const long nh = (long)GN_NODES * GHID;
  const long scatterTotal = (long)GN_EDGES * GHID;
  for (int i = 0; i < 2; ++i) {
    zero4_kernel<<<(int)((nh / 4 + 255) / 256), 256, 0, stream>>>((float4*)oth, nh / 4);
    scatter_kernel<<<(int)((scatterTotal + 255) / 256), 256, 0, stream>>>(src, dst, cur, oth,
                                                                          scatterTotal);
    layer_kernel<<<nodeBlocks, 256, 0, stream>>>(oth, cur,
                                                 lwT + (size_t)i * GHID * GHID,
                                                 conv_lb + (size_t)i * GHID,
                                                 rwT + (size_t)i * GHID * GHID,
                                                 invdeg, nodeTiles);
    float* t = cur; cur = oth; oth = t;
  }
  // cur now holds final h

  // edge head + mean
  zero_kernel<<<1, 32, 0, stream>>>(pathSum, 1L);
  edge_head_kernel<<<eBlocks, 256, 0, stream>>>(src, dst, cur, w1T, eh_b1, eh_w2, eh_b2,
                                                scores, pathSum, eTiles);

  // emit h
  copy_kernel<<<(int)((nh + 255) / 256), 256, 0, stream>>>(cur, hOut, nh);
}
